// BasicBlock_1w1a_45346264711393
// MI455X (gfx1250) — matrix operand order repaired and checked by reading, not compiled, not run on hardware
//
#include <hip/hip_runtime.h>
#include <hip/hip_bf16.h>

typedef __attribute__((ext_vector_type(16))) _Float16 v16h;
typedef __attribute__((ext_vector_type(8)))  _Float16 v8h;
typedef __attribute__((ext_vector_type(8)))  float    v8f;

#define BATCH 32
#define CH    256
#define HH    28
#define WW    28
#define HWSZ  (HH*WW)          // 784
#define MTOT  (BATCH*HWSZ)     // 25088
#define BN_EPS_F 1e-5f

#define BM   128
#define BNT  128
#define LDA  40                // 32 halves + 8 pad (80B row stride -> conflict-light)

// ---------------------------------------------------------------------------
// Kernel 1: NCHW -> NHWC transpose of x, also emit binarized activation (f16)
// ---------------------------------------------------------------------------
__global__ void k_transpose_binact(const float* __restrict__ x,
                                   const float* __restrict__ alpha_p,
                                   const float* __restrict__ beta_p,
                                   float* __restrict__ xt,
                                   _Float16* __restrict__ abin)
{
  __shared__ float tile[32][33];
  const int b   = blockIdx.x;
  const int hwT = blockIdx.y;   // 25 tiles of 32 over 784
  const int cT  = blockIdx.z;   // 8 tiles of 32 over 256
  const int tx  = threadIdx.x;  // 32
  const int ty  = threadIdx.y;  // 8
  const float alpha = alpha_p[0], beta = beta_p[0];

  #pragma unroll
  for (int j = 0; j < 4; ++j) {           // read coalesced along hw (NCHW inner dim)
    const int c  = cT*32 + ty + j*8;
    const int hw = hwT*32 + tx;
    float v = 0.f;
    if (hw < HWSZ) v = x[((size_t)b*CH + c)*HWSZ + hw];
    tile[ty + j*8][tx] = v;               // tile[c_local][hw_local]
  }
  __syncthreads();
  #pragma unroll
  for (int j = 0; j < 4; ++j) {           // write coalesced along c (NHWC inner dim)
    const int hw = hwT*32 + ty + j*8;
    const int c  = cT*32 + tx;
    if (hw < HWSZ) {
      const float v = tile[tx][ty + j*8];
      const size_t o = ((size_t)b*HWSZ + hw)*CH + c;
      xt[o] = v;
      const float xs = (v - beta) / alpha;
      const float s  = (xs >= 0.f) ? 1.f : -1.f;
      abin[o] = (_Float16)(alpha * (s + beta));
    }
  }
}

// ---------------------------------------------------------------------------
// Kernel 2: binarize weights.  OIHW f32 -> WB[(kh*3+kw)][o][ci] f16
// ---------------------------------------------------------------------------
__global__ void k_wprep(const float* __restrict__ w, _Float16* __restrict__ WB)
{
  __shared__ float red[256];
  const int o = blockIdx.x;     // output channel
  const int t = threadIdx.x;    // input channel
  float wv[9];
  const size_t base = ((size_t)o*CH + t)*9;
  float s = 0.f;
  #pragma unroll
  for (int k = 0; k < 9; ++k) { wv[k] = w[base + k]; s += wv[k]; }
  red[t] = s; __syncthreads();
  for (int off = 128; off > 0; off >>= 1) {
    if (t < off) red[t] += red[t + off];
    __syncthreads();
  }
  const float mean = red[0] * (1.f/2304.f);
  __syncthreads();
  float s2 = 0.f;
  #pragma unroll
  for (int k = 0; k < 9; ++k) { const float d = wv[k] - mean; s2 += d*d; }
  red[t] = s2; __syncthreads();
  for (int off = 128; off > 0; off >>= 1) {
    if (t < off) red[t] += red[t + off];
    __syncthreads();
  }
  const float alpha = sqrtf(red[0] * (1.f/2304.f));
  #pragma unroll
  for (int k = 0; k < 9; ++k) {
    const float wn = (wv[k] - mean) / alpha;
    const float sg = (wn >= 0.f) ? 1.f : -1.f;
    WB[(size_t)k*CH*CH + (size_t)o*CH + t] = (_Float16)(sg*alpha + mean);
  }
}

// ---------------------------------------------------------------------------
// Kernel 3: implicit-GEMM 3x3 conv via WMMA f16 -> f32.
//   A: abin NHWC [m][ci] f16 (m = b*784 + h*28 + w),  B: WB[(kh3kw)][o][ci] f16
//   C: [m][o] f32.  Block tile 128x128, 8 waves, wave tile 32x64 (2x4 WMMA).
// ---------------------------------------------------------------------------
__global__ __launch_bounds__(256) void k_gemm(const _Float16* __restrict__ A,
                                              const _Float16* __restrict__ Bw,
                                              float* __restrict__ C)
{
  __shared__ __align__(16) _Float16 sA[BM  * LDA];
  __shared__ __align__(16) _Float16 sB[BNT * LDA];

  const int tid   = threadIdx.x;
  const int bm    = blockIdx.x;
  const int bn    = blockIdx.y;
  const int lane  = tid & 31;
  const int wave  = tid >> 5;
  const int waveM = wave & 3;      // 4 waves over M (32 rows each)
  const int waveN = wave >> 2;     // 2 waves over N (64 cols each)
  const int lm    = lane & 15;
  const int lh    = lane >> 4;

  // cooperative load coords: 2 threads per row, 16 halves (32B) each
  const int rowL = tid >> 1;           // 0..127
  const int seg  = (tid & 1) * 16;     // half-offset 0 or 16

  // decode output position for our A row
  const int m   = bm*BM + rowL;
  const int bIx = m / HWSZ;
  const int hw  = m - bIx*HWSZ;
  const int oh  = hw / WW;
  const int ow  = hw - oh*WW;

  v8f acc[2][4] = {};
  const v8h zero8 = {};

  for (int khkw = 0; khkw < 9; ++khkw) {
    const int kh = khkw / 3, kw = khkw - kh*3;
    const int ih = oh + kh - 1, iw = ow + kw - 1;
    const bool valid = ((unsigned)ih < (unsigned)HH) && ((unsigned)iw < (unsigned)WW);
    const _Float16* gA = A  + ((size_t)((bIx*HH + ih)*WW + iw))*CH + seg;
    const _Float16* gB = Bw + (size_t)khkw*CH*CH + (size_t)(bn*BNT + rowL)*CH + seg;

    for (int ci0 = 0; ci0 < CH; ci0 += 32) {
      // ---- global -> regs
      v8h a0 = zero8, a1 = zero8;
      if (valid) {
        a0 = *(const v8h*)(gA + ci0);
        a1 = *(const v8h*)(gA + ci0 + 8);
      }
      const v8h b0 = *(const v8h*)(gB + ci0);
      const v8h b1 = *(const v8h*)(gB + ci0 + 8);
      __builtin_prefetch((const void*)(gB + ci0 + 32), 0, 1);  // global_prefetch next chunk

      __syncthreads();                 // previous tile fully consumed
      *(v8h*)&sA[rowL*LDA + seg]     = a0;
      *(v8h*)&sA[rowL*LDA + seg + 8] = a1;
      *(v8h*)&sB[rowL*LDA + seg]     = b0;
      *(v8h*)&sB[rowL*LDA + seg + 8] = b1;
      __syncthreads();                 // tile visible

      // ---- fragments per ISA 7.12.2 (wave32 16-bit layouts)
      v16h aF[2], bF[4];
      #pragma unroll
      for (int i = 0; i < 2; ++i) {
        const int r = (waveM*32 + i*16 + lm) * LDA;
        const v8h lo = *(const v8h*)&sA[r + lh*8];        // K 0..7 / 8..15
        const v8h hi = *(const v8h*)&sA[r + 16 + lh*8];   // K 16..23 / 24..31
        aF[i] = __builtin_shufflevector(lo, hi, 0,1,2,3,4,5,6,7,8,9,10,11,12,13,14,15);
      }
      #pragma unroll
      for (int j = 0; j < 4; ++j) {
        const int r = (waveN*64 + j*16 + lm) * LDA + lh*8; // K 0..7,16..23 / 8..15,24..31, N = lane
        const v8h lo = *(const v8h*)&sB[r];
        const v8h hi = *(const v8h*)&sB[r + 16];
        bF[j] = __builtin_shufflevector(lo, hi, 0,1,2,3,4,5,6,7,8,9,10,11,12,13,14,15);
      }
      #pragma unroll
      for (int i = 0; i < 2; ++i)
        #pragma unroll
        for (int j = 0; j < 4; ++j)
          acc[i][j] = __builtin_amdgcn_wmma_f32_16x16x32_f16(
              false, aF[i], false, bF[j], (short)0, acc[i][j], false, false);
    }
  }

  // ---- epilogue: D layout -> C[m][o]
  const int mBase = bm*BM  + waveM*32;
  const int nBase = bn*BNT + waveN*64;
  #pragma unroll
  for (int i = 0; i < 2; ++i)
    #pragma unroll
    for (int j = 0; j < 4; ++j) {
      const int col = nBase + j*16 + lm;
      #pragma unroll
      for (int r = 0; r < 8; ++r) {
        const int row = mBase + i*16 + lh*8 + r;
        C[(size_t)row*CH + col] = acc[i][j][r];
      }
    }
}

// ---------------------------------------------------------------------------
// Kernel 4: per-channel BN statistics (sum, sumsq) with float atomics
// ---------------------------------------------------------------------------
__global__ void k_bnstats(const float* __restrict__ C, float* __restrict__ stats)
{
  const int t  = threadIdx.x;      // channel
  const int r0 = blockIdx.x * 256; // row chunk
  float s = 0.f, s2 = 0.f;
  for (int r = 0; r < 256; ++r) {
    const float v = C[(size_t)(r0 + r)*CH + t];
    s  += v;
    s2 += v*v;
  }
  atomicAdd(&stats[t],      s);
  atomicAdd(&stats[CH + t], s2);
}

// ---------------------------------------------------------------------------
// Kernel 5: fused BN + residual + maxout (+ optional next-layer binarization)
// ---------------------------------------------------------------------------
__global__ void k_fuse(const float* __restrict__ C, const float* __restrict__ stats,
                       const float* __restrict__ g, const float* __restrict__ bb,
                       const float* __restrict__ pos, const float* __restrict__ neg,
                       const float* __restrict__ res,
                       float* __restrict__ y,
                       _Float16* __restrict__ abin,
                       const float* __restrict__ alpha_p,
                       const float* __restrict__ beta_p)
{
  const int idx = blockIdx.x*256 + threadIdx.x;
  const int c   = idx & 255;
  const float mean = stats[c]      * (1.f/(float)MTOT);
  const float var  = stats[CH + c] * (1.f/(float)MTOT) - mean*mean;
  const float sc   = g[c] * rsqrtf(var + BN_EPS_F);
  const float sh   = bb[c] - mean*sc;
  const float v    = C[idx]*sc + sh + res[idx];
  const float o    = (v > 0.f) ? pos[c]*v : neg[c]*v;
  y[idx] = o;
  if (abin) {
    const float alpha = alpha_p[0], beta = beta_p[0];
    const float xs = (o - beta) / alpha;
    const float s  = (xs >= 0.f) ? 1.f : -1.f;
    abin[idx] = (_Float16)(alpha * (s + beta));
  }
}

// ---------------------------------------------------------------------------
// Kernel 6: NHWC -> NCHW transpose into d_out
// ---------------------------------------------------------------------------
__global__ void k_transpose_out(const float* __restrict__ y, float* __restrict__ out)
{
  __shared__ float tile[32][33];
  const int b   = blockIdx.x;
  const int hwT = blockIdx.y;
  const int cT  = blockIdx.z;
  const int tx  = threadIdx.x;
  const int ty  = threadIdx.y;
  #pragma unroll
  for (int j = 0; j < 4; ++j) {           // read coalesced along c
    const int hw = hwT*32 + ty + j*8;
    const int c  = cT*32 + tx;
    float v = 0.f;
    if (hw < HWSZ) v = y[((size_t)b*HWSZ + hw)*CH + c];
    tile[ty + j*8][tx] = v;               // tile[hw_local][c_local]
  }
  __syncthreads();
  #pragma unroll
  for (int j = 0; j < 4; ++j) {           // write coalesced along hw
    const int hw = hwT*32 + tx;
    const int c  = cT*32 + ty + j*8;
    if (hw < HWSZ) out[((size_t)b*CH + c)*HWSZ + hw] = tile[tx][ty + j*8];
  }
}

// ---------------------------------------------------------------------------
extern "C" void kernel_launch(void* const* d_in, const int* in_sizes, int n_in,
                              void* d_out, int out_size, void* d_ws, size_t ws_size,
                              hipStream_t stream)
{
  (void)in_sizes; (void)n_in; (void)out_size; (void)ws_size;
  const float* x   = (const float*)d_in[0];
  const float* aa1 = (const float*)d_in[1];
  const float* ba1 = (const float*)d_in[2];
  const float* w1  = (const float*)d_in[3];
  const float* g1  = (const float*)d_in[4];
  const float* b1  = (const float*)d_in[5];
  const float* p1  = (const float*)d_in[6];
  const float* n1  = (const float*)d_in[7];
  const float* aa2 = (const float*)d_in[8];
  const float* ba2 = (const float*)d_in[9];
  const float* w2  = (const float*)d_in[10];
  const float* g2  = (const float*)d_in[11];
  const float* b2  = (const float*)d_in[12];
  const float* p2  = (const float*)d_in[13];
  const float* n2  = (const float*)d_in[14];
  float* out = (float*)d_out;

  const size_t N = (size_t)MTOT * CH;           // 6,422,528 elems
  char* ws = (char*)d_ws;
  float*    xt    = (float*)ws;    ws += N*4;   // NHWC copy of x (later reused for y2)
  float*    x1    = (float*)ws;    ws += N*4;   // layer-1 output (NHWC)
  float*    conv  = (float*)ws;    ws += N*4;   // GEMM output [m][o]
  _Float16* abin  = (_Float16*)ws; ws += N*2;   // binarized activations (NHWC f16)
  _Float16* WB    = (_Float16*)ws; ws += (size_t)9*CH*CH*2; // binarized weights
  float*    stats = (float*)ws;    ws += 2*CH*4;

  const dim3 tgrid(BATCH, 25, 8), tblk(32, 8);
  const dim3 ggrid(MTOT/BM, CH/BNT);            // (196, 2)

  // ---- layer 1
  k_transpose_binact<<<tgrid, tblk, 0, stream>>>(x, aa1, ba1, xt, abin);
  k_wprep<<<CH, 256, 0, stream>>>(w1, WB);
  k_gemm<<<ggrid, 256, 0, stream>>>(abin, WB, conv);
  hipMemsetAsync(stats, 0, 2*CH*sizeof(float), stream);
  k_bnstats<<<MTOT/256, 256, 0, stream>>>(conv, stats);
  k_fuse<<<(int)(N/256), 256, 0, stream>>>(conv, stats, g1, b1, p1, n1,
                                           xt, x1, abin, aa2, ba2);
  // ---- layer 2
  k_wprep<<<CH, 256, 0, stream>>>(w2, WB);
  k_gemm<<<ggrid, 256, 0, stream>>>(abin, WB, conv);
  hipMemsetAsync(stats, 0, 2*CH*sizeof(float), stream);
  k_bnstats<<<MTOT/256, 256, 0, stream>>>(conv, stats);
  k_fuse<<<(int)(N/256), 256, 0, stream>>>(conv, stats, g2, b2, p2, n2,
                                           x1, xt, (_Float16*)nullptr, aa2, ba2);
  // ---- NHWC -> NCHW into d_out
  k_transpose_out<<<tgrid, tblk, 0, stream>>>(xt, out);
}
